// SineSPE_52218212385266
// MI455X (gfx1250) — matrix operand order repaired and checked by reading, not compiled, not run on hardware
//
#include <hip/hip_runtime.h>
#include <math.h>

#define B_ 2
#define L_ 1024
#define H_ 8
#define D_ 64
#define S_ 10
#define R_ 128
#define K_ (D_ * 2 * S_)   // 1280
#define KB 32              // K-block per WMMA sweep
#define NBLK (K_ / KB)     // 40
#define MT 128             // L-rows per workgroup
#define THREADS 512        // 16 wave32s

typedef _Float16 v8h  __attribute__((ext_vector_type(8)));
typedef _Float16 v16h __attribute__((ext_vector_type(16)));
typedef float    v8f  __attribute__((ext_vector_type(8)));

union H2U { unsigned u; _Float16 h[2]; };

static __device__ __forceinline__ unsigned pack2h(float a, float b) {
    H2U v; v.h[0] = (_Float16)a; v.h[1] = (_Float16)b; return v.u;
}

// ---------------------------------------------------------------------------
// Pass 1: zsT[b,h,r,k] = z[b,h,d,t,r] * softplus(gains[h,d,t%S]) / sqrt(R*D)
//         stored f16, transposed so GEMM B-fragments are contiguous in k.
// ---------------------------------------------------------------------------
__global__ void spe_prep_z(const float* __restrict__ z,
                           const float* __restrict__ gains,
                           _Float16* __restrict__ zsT) {
    int idx = blockIdx.x * blockDim.x + threadIdx.x;   // linear over z [B,H,D,2S,R]
    if (idx >= B_ * H_ * D_ * 2 * S_ * R_) return;
    int r = idx % R_;
    int t = (idx / R_) % (2 * S_);
    int d = (idx / (R_ * 2 * S_)) % D_;
    int h = (idx / (R_ * 2 * S_ * D_)) % H_;
    int b = idx / (R_ * 2 * S_ * D_ * H_);
    float g = gains[(h * D_ + d) * S_ + (t % S_)];
    float sp = (g > 0.f) ? (g + log1pf(__expf(-g))) : log1pf(__expf(g));  // softplus
    const float scale = 0.011048543456039806f;          // 1/sqrt(128*64)
    float v = z[idx] * sp * scale;
    int k = d * 2 * S_ + t;
    zsT[(size_t)((b * H_ + h) * R_ + r) * K_ + k] = (_Float16)v;
}

// ---------------------------------------------------------------------------
// Pass 2: omega[h,l, d*2S + 2s(+1)] = cos/sin(2*pi * sigmoid(freqs)/2 * l), f16
// ---------------------------------------------------------------------------
__global__ void spe_prep_omega(const float* __restrict__ freqs,
                               _Float16* __restrict__ omega) {
    int idx = blockIdx.x * blockDim.x + threadIdx.x;   // over H*L*D*S
    if (idx >= H_ * L_ * D_ * S_) return;
    int s = idx % S_;
    int d = (idx / S_) % D_;
    int l = (idx / (S_ * D_)) % L_;
    int h = idx / (S_ * D_ * L_);
    float f = freqs[(h * D_ + d) * S_ + s];
    float fr = 0.5f / (1.f + __expf(-f));               // sigmoid/2
    float ph = 6.283185307179586f * fr * (float)l;
    float sn, cs;
    __sincosf(ph, &sn, &cs);
    size_t e = ((size_t)h * L_ + l) * K_ + d * 2 * S_ + 2 * s;  // even -> 4B aligned
    *(unsigned*)(omega + e) = pack2h(cs, sn);
}

// ---------------------------------------------------------------------------
// Pass 3: batched GEMM with on-the-fly A construction.
//   grid.x = b*H+h (16), grid.y = L/MT (8); 512 threads = 16 waves.
//   B tiles streamed global->LDS with GLOBAL_LOAD_ASYNC_TO_LDS_B128, double
//   buffered so the copy overlaps A-tile VALU construction and WMMA.
// ---------------------------------------------------------------------------
__global__ __launch_bounds__(THREADS) void spe_gemm(
    const float* __restrict__ queries, const float* __restrict__ keys,
    const float* __restrict__ offsets,
    const _Float16* __restrict__ omega, const _Float16* __restrict__ zsT,
    float* __restrict__ out) {
    __shared__ float  q_lds[MT * D_];                       // 32 KB
    __shared__ float  k_lds[MT * D_];                       // 32 KB
    __shared__ float2 oc_lds[D_ * S_];                      // 5 KB  (cos,sin of offsets)
    __shared__ __align__(32) _Float16 Aq_lds[MT * KB];      // 8 KB
    __shared__ __align__(32) _Float16 Ak_lds[MT * KB];      // 8 KB
    __shared__ __align__(32) _Float16 Bt0_lds[R_ * KB];     // 8 KB (B double buffer 0)
    __shared__ __align__(32) _Float16 Bt1_lds[R_ * KB];     // 8 KB (B double buffer 1)

    const int tid = threadIdx.x;
    const int bh  = blockIdx.x;
    const int b   = bh / H_;
    const int h   = bh % H_;
    const int l0  = blockIdx.y * MT;

    const _Float16* omh = omega + (size_t)h * L_ * K_;
    const _Float16* zb  = zsT + (size_t)bh * R_ * K_;

    // Per-thread async-copy slot: one b128 each covers the whole 8 KB B tile.
    const int brow = tid >> 2;             // 0..127
    const int bcol = (tid & 3) * 8;        // f16 offset 0,8,16,24 (16B chunks)
    const _Float16* bsrc = zb + (size_t)brow * K_ + bcol;
    unsigned bdst0 = (unsigned)(size_t)(Bt0_lds + brow * KB + bcol);
    unsigned bdst1 = (unsigned)(size_t)(Bt1_lds + brow * KB + bcol);

    // Kick off async copy of B tile for block 0 (ASYNCcnt path, no VGPR round-trip).
    asm volatile("global_load_async_to_lds_b128 %0, %1, off"
                 :: "v"(bdst0), "v"(bsrc) : "memory");

    // Stage query/key rows for this L-tile.
    for (int e = tid; e < MT * D_; e += THREADS) {
        int row = e / D_, d = e % D_;
        size_t gi = ((size_t)(b * L_ + l0 + row) * H_ + h) * D_ + d;
        q_lds[e] = queries[gi];
        k_lds[e] = keys[gi];
    }
    // Offset sincos table for this head.
    for (int e = tid; e < D_ * S_; e += THREADS) {
        float o = offsets[h * D_ * S_ + e];
        float sn, cs;
        __sincosf(o, &sn, &cs);
        oc_lds[e] = make_float2(cs, sn);
    }
    __syncthreads();

    const int wave   = tid >> 5;
    const int lane   = tid & 31;
    const int l15    = lane & 15;
    const int hi     = (lane >= 16);           // K-half select per WMMA layout
    const int mrow   = (wave & 7) * 16 + l15;  // A-frag row within tile
    const int ntbase = (wave >> 3) * 4;        // n-tile group: 0 or 4

    v8f accq[4] = {};
    v8f acck[4] = {};

    for (int blk = 0; blk < NBLK; ++blk) {
        const int kb0 = blk * KB;
        const bool has_next = (blk + 1 < NBLK);

        // Issue async copy of NEXT B tile into the idle buffer; it overlaps the
        // A-tile VALU construction and this block's WMMAs.
        if (has_next) {
            unsigned nd = (blk & 1) ? bdst0 : bdst1;
            asm volatile("global_load_async_to_lds_b128 %0, %1, off"
                         :: "v"(nd), "v"(bsrc + kb0 + KB) : "memory");
        }
        // Prefetch next omega rows while we are at it.
        if (has_next) {
            for (int n = tid; n < MT; n += THREADS)
                __builtin_prefetch(omh + (size_t)(l0 + n) * K_ + kb0 + KB, 0, 0);
        }

        // ---- A tiles: pair task = (row, cos/sin pair) ----
        for (int tk = tid; tk < MT * (KB / 2); tk += THREADS) {
            int row = tk / (KB / 2);
            int p   = tk % (KB / 2);
            int kk  = kb0 + 2 * p;
            int d   = kk / (2 * S_);
            int s2  = (kk % (2 * S_)) >> 1;
            H2U cs; cs.u = *(const unsigned*)(omh + (size_t)(l0 + row) * K_ + kk);
            float c = (float)cs.h[0], s = (float)cs.h[1];
            float qv = q_lds[row * D_ + d];
            float kv = k_lds[row * D_ + d];
            ((unsigned*)Ak_lds)[tk] = pack2h(c * kv, s * kv);
            float2 oc = oc_lds[d * S_ + s2];
            float cq = c * oc.x - s * oc.y;     // cos(x+off)
            float sq = s * oc.x + c * oc.y;     // sin(x+off)
            ((unsigned*)Aq_lds)[tk] = pack2h(cq * qv, sq * qv);
        }

        // Current block's async copy must have landed. Same-type async ops
        // complete in order, so <=1 outstanding means block `blk` is done.
        if (has_next)
            asm volatile("s_wait_asynccnt 0x1" ::: "memory");
        else
            asm volatile("s_wait_asynccnt 0x0" ::: "memory");
        __syncthreads();

        const _Float16* bt = (blk & 1) ? Bt1_lds : Bt0_lds;

        // ---- fragments + WMMA ----
        // A 16x32 f16, contiguous k order chosen here (B uses the same): lanes 0-15 row m hold K 0-15; lanes 16-31 hold K 16-31.
        const _Float16* arq = Aq_lds + mrow * KB;
        const _Float16* ark = Ak_lds + mrow * KB;
        int ka = hi ? 16 : 0;
        v8h aq0 = *(const v8h*)(arq + ka);
        v8h aq1 = *(const v8h*)(arq + ka + 8);
        v8h ak0 = *(const v8h*)(ark + ka);
        v8h ak1 = *(const v8h*)(ark + ka + 8);
        v16h aq = __builtin_shufflevector(aq0, aq1, 0,1,2,3,4,5,6,7,8,9,10,11,12,13,14,15);
        v16h ak = __builtin_shufflevector(ak0, ak1, 0,1,2,3,4,5,6,7,8,9,10,11,12,13,14,15);

#pragma unroll
        for (int nt = 0; nt < 4; ++nt) {
            // B 32x16, same contiguous k order as A: lanes 0-15 col n hold K 0-15; lanes 16-31 hold K 16-31.
            int n = (ntbase + nt) * 16 + l15;
            v16h bf = *(const v16h*)(bt + n * KB + (hi ? 16 : 0));
            accq[nt] = __builtin_amdgcn_wmma_f32_16x16x32_f16(
                false, aq, false, bf, (short)0, accq[nt], false, false);
            acck[nt] = __builtin_amdgcn_wmma_f32_16x16x32_f16(
                false, ak, false, bf, (short)0, acck[nt], false, false);
        }
        __syncthreads();
    }

    // ---- epilogue: C/D layout — VGPR j: lanes 0-15 M=j, lanes 16-31 M=j+8; N=lane&15 ----
    const size_t outK = (size_t)B_ * L_ * H_ * R_;
#pragma unroll
    for (int nt = 0; nt < 4; ++nt) {
        int r = (ntbase + nt) * 16 + l15;
#pragma unroll
        for (int j = 0; j < 8; ++j) {
            int m = hi ? (8 + j) : j;
            int l = l0 + (wave & 7) * 16 + m;
            size_t oi = ((size_t)(b * L_ + l) * H_ + h) * R_ + r;
            out[oi]        = accq[nt][j];
            out[outK + oi] = acck[nt][j];
        }
    }
}

// ---------------------------------------------------------------------------
extern "C" void kernel_launch(void* const* d_in, const int* in_sizes, int n_in,
                              void* d_out, int out_size, void* d_ws, size_t ws_size,
                              hipStream_t stream) {
    const float* queries = (const float*)d_in[0];
    const float* keys    = (const float*)d_in[1];
    const float* freqs   = (const float*)d_in[2];
    const float* offsets = (const float*)d_in[3];
    const float* gains   = (const float*)d_in[4];
    const float* z       = (const float*)d_in[5];

    _Float16* omega = (_Float16*)d_ws;                       // H*L*K f16 = 21 MB
    _Float16* zsT   = omega + (size_t)H_ * L_ * K_;          // B*H*R*K f16 = 5.2 MB

    int nz = B_ * H_ * D_ * 2 * S_ * R_;
    spe_prep_z<<<(nz + 255) / 256, 256, 0, stream>>>(z, gains, zsT);

    int no = H_ * L_ * D_ * S_;
    spe_prep_omega<<<(no + 255) / 256, 256, 0, stream>>>(freqs, omega);

    dim3 grid(B_ * H_, L_ / MT);
    spe_gemm<<<grid, THREADS, 0, stream>>>(queries, keys, offsets, omega, zsT,
                                           (float*)d_out);
}
